// GCNNodeEncoder_44023414784042
// MI455X (gfx1250) — matrix operand order repaired and checked by reading, not compiled, not run on hardware
//
#include <hip/hip_runtime.h>
#include <stdint.h>

#define N_FEAT   128
#define HID      16
#define LN_EPS   1e-5f
#define P_DROP   0.3f
#define WAVES_PB 8   // waves per block (256 threads)

typedef __attribute__((ext_vector_type(16))) __bf16 v16bf;
typedef __attribute__((ext_vector_type(8)))  __bf16 v8bf;
typedef __attribute__((ext_vector_type(8)))  float  v8f;

__device__ __forceinline__ float hash_u01(uint32_t x) {
    x ^= x >> 16; x *= 0x7feb352dU;
    x ^= x >> 15; x *= 0x846ca68bU;
    x ^= x >> 16;
    return (float)(x >> 8) * (1.0f / 16777216.0f);
}

__device__ __forceinline__ void atomic_add_f32(float* p, float v) {
    __hip_atomic_fetch_add(p, v, __ATOMIC_RELAXED, __HIP_MEMORY_SCOPE_AGENT);
}

// ---------------- init: deg = 1.0 (self loops), zero both accumulators ----
__global__ void k_init(float* deg, float* accB, float* accC, int n_nodes) {
    long tid = (long)blockIdx.x * blockDim.x + threadIdx.x;
    long total = (long)n_nodes * HID;
    if (tid < n_nodes) deg[tid] = 1.0f;
    if (tid < total) { accB[tid] = 0.0f; accC[tid] = 0.0f; }
}

// ---------------- degree accumulation over dst ----------------------------
__global__ void k_deg(const int* __restrict__ dst, float* deg, int n_edges) {
    int e = blockIdx.x * blockDim.x + threadIdx.x;
    if (e >= n_edges) return;
    atomic_add_f32(&deg[dst[e]], 1.0f);
}

__global__ void k_dinv(float* deg, int n_nodes) {
    int i = blockIdx.x * blockDim.x + threadIdx.x;
    if (i >= n_nodes) return;
    deg[i] = rsqrtf(deg[i]);       // in place: deg -> deg^{-1/2}
}

// ---------------- fused LayerNorm + (xn @ W1) via WMMA bf16 ---------------
// one wave = one 16-node tile; 4x V_WMMA_F32_16X16X32_BF16 over K=128.
// B fragments (shared by all waves) are pre-swizzled into lane layout in LDS.
__global__ void __launch_bounds__(256)
k_ln_gemm1(const float* __restrict__ x, const float* __restrict__ gamma,
           const float* __restrict__ beta, const float* __restrict__ W1,
           float* __restrict__ hlin, int n_nodes) {
    __shared__ __align__(16) __bf16 lx[WAVES_PB][16][N_FEAT]; // normalized rows (32 KB)
    __shared__ __align__(32) __bf16 lwB[4][32][16];           // B fragments (4 KB)

    const int t = threadIdx.x, w = t >> 5, lane = t & 31;

    // stage W1 directly in B-fragment lane layout:
    // lane ln holds column n=ln&15; element e is K = kk*32 + (e>>3)*16 + (ln>>4)*8 + (e&7)
    for (int i = t; i < 4 * 32 * 16; i += 256) {
        const int kk = i >> 9;
        const int ln = (i >> 4) & 31;
        const int e  = i & 15;
        const int nn = ln & 15, hh = ln >> 4;
        lwB[kk][ln][e] = (__bf16)W1[(kk * 32 + (e >> 3) * 16 + hh * 8 + (e & 7)) * HID + nn];
    }

    const int ntiles = n_nodes >> 4;               // N is a multiple of 16
    const int tile = blockIdx.x * WAVES_PB + w;
    const bool valid = tile < ntiles;

    if (valid) {
        // LayerNorm: 2 lanes per row, 64 features per lane
        const int r = lane >> 1, half = lane & 1;
        const long node = (long)tile * 16 + r;
        const float4* xr = (const float4*)(x + node * N_FEAT + half * 64);
        float4 v[16];
        float s = 0.f, q = 0.f;
#pragma unroll
        for (int j = 0; j < 16; ++j) {
            v[j] = xr[j];
            s += v[j].x + v[j].y + v[j].z + v[j].w;
            q += v[j].x * v[j].x + v[j].y * v[j].y + v[j].z * v[j].z + v[j].w * v[j].w;
        }
        s += __shfl_xor(s, 1, 32);
        q += __shfl_xor(q, 1, 32);
        const float mean = s * (1.0f / N_FEAT);
        const float var  = q * (1.0f / N_FEAT) - mean * mean;
        const float rs   = rsqrtf(var + LN_EPS);
#pragma unroll
        for (int j = 0; j < 16; ++j) {
            const int c = half * 64 + j * 4;
            float4 g  = *(const float4*)&gamma[c];
            float4 bb = *(const float4*)&beta[c];
            lx[w][r][c + 0] = (__bf16)((v[j].x - mean) * rs * g.x + bb.x);
            lx[w][r][c + 1] = (__bf16)((v[j].y - mean) * rs * g.y + bb.y);
            lx[w][r][c + 2] = (__bf16)((v[j].z - mean) * rs * g.z + bb.z);
            lx[w][r][c + 3] = (__bf16)((v[j].w - mean) * rs * g.w + bb.w);
        }
    }
    __syncthreads();

    if (valid) {
        const int m = lane & 15, h = lane >> 4, n = lane & 15;
        v8f c = {0.f, 0.f, 0.f, 0.f, 0.f, 0.f, 0.f, 0.f};
#pragma unroll
        for (int kk = 0; kk < 4; ++kk) {
            const int k0 = kk * 32;
            // A fragment: lanes 0-15 K=k0+0..7 / k0+16..23, lanes 16-31 +8
            v8bf lo = *(const v8bf*)&lx[w][m][k0 + h * 8];
            v8bf hi = *(const v8bf*)&lx[w][m][k0 + 16 + h * 8];
            v16bf a;
#pragma unroll
            for (int i = 0; i < 8; ++i) { a[i] = lo[i]; a[8 + i] = hi[i]; }
            // B fragment: contiguous pre-swizzled 32B per lane
            v16bf b = *(const v16bf*)&lwB[kk][lane][0];
            c = __builtin_amdgcn_wmma_f32_16x16x32_bf16(
                    false, a, false, b, (short)0, c, false, false);
        }
        const long base = (long)tile * 16;
#pragma unroll
        for (int vv = 0; vv < 8; ++vv)
            hlin[(base + vv + 8 * h) * HID + n] = c[vv];
    }
}

// ---------------- edge scatter: acc[dst] += hlin[src] * dinv_s*dinv_d -----
__global__ void k_scatter(const int* __restrict__ src, const int* __restrict__ dst,
                          const float* __restrict__ dinv, const float* __restrict__ hlin,
                          float* acc, int n_edges) {
    int e = blockIdx.x * blockDim.x + threadIdx.x;
    if (e >= n_edges) return;
    const int s = src[e], d = dst[e];
    const float wgt = dinv[s] * dinv[d];
    const float4* hs = (const float4*)&hlin[(long)s * HID];
    float* ap = &acc[(long)d * HID];
#pragma unroll
    for (int j = 0; j < 4; ++j) {
        float4 hv = hs[j];
        atomic_add_f32(ap + j * 4 + 0, hv.x * wgt);
        atomic_add_f32(ap + j * 4 + 1, hv.y * wgt);
        atomic_add_f32(ap + j * 4 + 2, hv.z * wgt);
        atomic_add_f32(ap + j * 4 + 3, hv.w * wgt);
    }
}

// ---------------- self-loop + bias + relu + dropout -----------------------
__global__ void k_pointwise(const float* __restrict__ acc, const float* __restrict__ hlin,
                            const float* __restrict__ dinv, const float* __restrict__ bias,
                            float* out, int n_nodes, uint32_t seed) {
    long tid = (long)blockIdx.x * blockDim.x + threadIdx.x;
    if (tid >= (long)n_nodes * HID) return;
    const int i = (int)(tid >> 4), f = (int)(tid & 15);
    const float di = dinv[i];
    float v = acc[tid] + hlin[tid] * di * di + bias[f];
    v = fmaxf(v, 0.0f);
    const float u = hash_u01((uint32_t)tid ^ seed);
    out[tid] = (u >= P_DROP) ? v * (1.0f / (1.0f - P_DROP)) : 0.0f;
}

// ---------------- h1 @ W2 via one WMMA (K=16 zero-padded to 32) -----------
__global__ void __launch_bounds__(256)
k_gemm2(const float* __restrict__ h1, const float* __restrict__ W2,
        float* __restrict__ hlin2, int n_nodes) {
    __shared__ __align__(32) __bf16 lwB2[32][16];  // B fragments (1 KB)

    const int t = threadIdx.x, w = t >> 5, lane = t & 31;

    // B: lane half hh holds K=hh*8+e for e<8; slots e>=8 (K>=16) -> zero
    for (int i = t; i < 32 * 16; i += 256) {
        const int ln = i >> 4, e = i & 15;
        const int nn = ln & 15, hh = ln >> 4;
        lwB2[ln][e] = (e < 8) ? (__bf16)W2[(hh * 8 + e) * HID + nn] : (__bf16)0.0f;
    }
    __syncthreads();

    const int tile = blockIdx.x * WAVES_PB + w;
    if (tile >= (n_nodes >> 4)) return;            // wave-uniform exit
    const int m = lane & 15, h = lane >> 4, n = lane & 15;

    // A: lane(m,h) holds K = h*8 + e for e<8 ; K>=16 slots are zero
    const float* hr = &h1[((long)tile * 16 + m) * HID + h * 8];
    float4 p0 = *(const float4*)hr;
    float4 p1 = *(const float4*)(hr + 4);
    v16bf a;
    a[0] = (__bf16)p0.x; a[1] = (__bf16)p0.y; a[2] = (__bf16)p0.z; a[3] = (__bf16)p0.w;
    a[4] = (__bf16)p1.x; a[5] = (__bf16)p1.y; a[6] = (__bf16)p1.z; a[7] = (__bf16)p1.w;
#pragma unroll
    for (int i = 8; i < 16; ++i) a[i] = (__bf16)0.0f;
    v16bf b = *(const v16bf*)&lwB2[lane][0];

    v8f c = {0.f, 0.f, 0.f, 0.f, 0.f, 0.f, 0.f, 0.f};
    c = __builtin_amdgcn_wmma_f32_16x16x32_bf16(false, a, false, b, (short)0, c,
                                                false, false);
    const long base = (long)tile * 16;
#pragma unroll
    for (int vv = 0; vv < 8; ++vv)
        hlin2[(base + vv + 8 * h) * HID + n] = c[vv];
}

extern "C" void kernel_launch(void* const* d_in, const int* in_sizes, int n_in,
                              void* d_out, int out_size, void* d_ws, size_t ws_size,
                              hipStream_t stream) {
    const float* x     = (const float*)d_in[0];
    const int*   eidx  = (const int*)d_in[1];
    const float* gamma = (const float*)d_in[2];
    const float* beta  = (const float*)d_in[3];
    const float* W1    = (const float*)d_in[4];
    const float* b1    = (const float*)d_in[5];
    const float* W2    = (const float*)d_in[6];
    const float* b2    = (const float*)d_in[7];
    float* out = (float*)d_out;

    const int N = in_sizes[0] / N_FEAT;     // 100000 (multiple of 16)
    const int E = in_sizes[1] / 2;          // 6400000
    const int* src = eidx;
    const int* dst = eidx + E;

    // workspace layout (fp32): dinv[N] | bufA[N*16] | bufB[N*16] | bufC[N*16]
    float* dinv = (float*)d_ws;
    float* bufA = dinv + N;                 // hlin1, later hlin2
    float* bufB = bufA + (long)N * HID;     // layer-1 acc -> h1
    float* bufC = bufB + (long)N * HID;     // layer-2 acc

    const int TPB = 256;
    const long nfTot = (long)N * HID;
    const int gNF = (int)((nfTot + TPB - 1) / TPB);
    const int gE  = (E + TPB - 1) / TPB;
    const int gN  = (N + TPB - 1) / TPB;
    const int ntiles = N / 16;
    const int gT  = (ntiles + WAVES_PB - 1) / WAVES_PB;

    k_init<<<gNF, TPB, 0, stream>>>(dinv, bufB, bufC, N);
    k_deg<<<gE, TPB, 0, stream>>>(dst, dinv, E);
    k_dinv<<<gN, TPB, 0, stream>>>(dinv, N);
    k_ln_gemm1<<<gT, TPB, 0, stream>>>(x, gamma, beta, W1, bufA, N);
    k_scatter<<<gE, TPB, 0, stream>>>(src, dst, dinv, bufA, bufB, E);
    k_pointwise<<<gNF, TPB, 0, stream>>>(bufB, bufA, dinv, b1, bufB, N, 0x9e3779b9u);
    k_gemm2<<<gT, TPB, 0, stream>>>(bufB, W2, bufA, N);
    k_scatter<<<gE, TPB, 0, stream>>>(src, dst, dinv, bufA, bufC, E);
    k_pointwise<<<gNF, TPB, 0, stream>>>(bufC, bufA, dinv, b2, out, N, 0x85ebca6bu);
}
